// ExRestSelfAtten_13168369730242
// MI455X (gfx1250) — matrix operand order repaired and checked by reading, not compiled, not run on hardware
//
#include <hip/hip_runtime.h>

// ---------------------------------------------------------------------------
// ExRestSelfAtten for MI455X (gfx1250, wave32, WMMA).
//
// Pipeline (all big GEMMs via v_wmma_f32_16x16x32_f16):
//   h  = relu(x @ W1 + b1)          f16 out (hh)
//   Q  = h @ Wq, K = h @ Wk, V = h @ Wv   f32 out
//   scores[w] = <Q[s], K[s+A-w]> / sqrt(512), softmax over 11 (zeros for pad)
//   aw = sum_w p[w] * V[s+A-w]      f32 out (output #2) + f16 copy
//   g  = relu(aw @ W2 + b2)         f16 out (gh)
//   out = g @ Wfc + bfc             f32 (output #1)
// ---------------------------------------------------------------------------

typedef __attribute__((ext_vector_type(16))) _Float16 v16h;
typedef __attribute__((ext_vector_type(8)))  _Float16 v8h;
typedef __attribute__((ext_vector_type(8)))  float    v8f;

#define BATCH   8
#define SEQ     2048
#define HID_DIM 512
#define NROWS   (BATCH * SEQ)
#define WN      11
#define ASZ     5

// ---- f32 -> f16 elementwise convert --------------------------------------
__global__ void cvt_f16_kernel(const float* __restrict__ src,
                               _Float16* __restrict__ dst, int n) {
    int i = blockIdx.x * blockDim.x + threadIdx.x;
    if (i < n) dst[i] = (_Float16)src[i];
}

// ---- transpose + convert: WT[n*K + k] = (f16) W[k*N + n] -----------------
__global__ void transpose_f16_kernel(const float* __restrict__ W,
                                     _Float16* __restrict__ WT, int K, int N) {
    int i = blockIdx.x * blockDim.x + threadIdx.x;
    if (i < K * N) {
        int k = i / N;
        int n = i - k * N;
        WT[(size_t)n * K + k] = (_Float16)W[i];
    }
}

// ---- WMMA GEMM: C[M,512] = act(A[M,512] @ BT^T + bias) -------------------
// A row-major f16; BT is B transposed (BT[n][k]) so both operands stream
// contiguously along K.  One wave computes a 64x32 tile (4x2 WMMA tiles).
//
// Fragment layouts used here (wave32; contiguous k, not the ISA table order):
//   A 16x32 f16 : lane(r,hx): elem j   -> K=16*hx+j  (row m=r)  [two 16B loads]
//   B 32x16 f16 : lane(r,hx): elem j   -> K=16*hx+j  (col n=r)  [one 32B load]
//   C 16x16 f32 : vgpr i -> (m = 8*hx + i, n = r)
__global__ __launch_bounds__(32) void gemm_wmma_kernel(
    const _Float16* __restrict__ A,   // [M, 512]
    const _Float16* __restrict__ BT,  // [512, 512], BT[n][k]
    const float*    __restrict__ bias, // [512] or nullptr
    float*          __restrict__ Cf,   // f32 out or nullptr
    _Float16*       __restrict__ Ch,   // f16 out or nullptr
    int relu)
{
    const int K = HID_DIM, N = HID_DIM;
    const int lane = threadIdx.x & 31;
    const int hx   = lane >> 4;    // half of wave
    const int r    = lane & 15;    // row (A) / col (B,C)
    const int n0 = blockIdx.x * 32;
    const int m0 = blockIdx.y * 64;

    union V16 { v16h v; v8h h[2]; };

    v8f acc[4][2] = {};

    for (int k0 = 0; k0 < K; k0 += 32) {
        V16 a[4];
        v16h b[2];
#pragma unroll
        for (int t = 0; t < 4; ++t) {
            const _Float16* ap = A + (size_t)(m0 + 16 * t + r) * K + (k0 + hx * 16);
            a[t].h[0] = *(const v8h*)(ap);        // K = k0 + 16*hx + 0..7
            a[t].h[1] = *(const v8h*)(ap + 8);    // K = k0 + 16*hx + 8..15
        }
#pragma unroll
        for (int t = 0; t < 2; ++t) {
            const _Float16* bp = BT + (size_t)(n0 + 16 * t + r) * K + (k0 + hx * 16);
            b[t] = *(const v16h*)bp;              // K = k0 + 16*hx + 0..15
        }
#pragma unroll
        for (int mi = 0; mi < 4; ++mi)
#pragma unroll
            for (int ni = 0; ni < 2; ++ni)
                acc[mi][ni] = __builtin_amdgcn_wmma_f32_16x16x32_f16(
                    false, a[mi].v, false, b[ni], (short)0, acc[mi][ni],
                    false, false);
    }

#pragma unroll
    for (int mi = 0; mi < 4; ++mi) {
#pragma unroll
        for (int ni = 0; ni < 2; ++ni) {
#pragma unroll
            for (int i = 0; i < 8; ++i) {
                int m = m0 + 16 * mi + 8 * hx + i;
                int n = n0 + 16 * ni + r;
                float v = acc[mi][ni][i];
                if (bias) v += bias[n];
                if (relu) v = fmaxf(v, 0.0f);
                if (Cf) Cf[(size_t)m * N + n] = v;
                if (Ch) Ch[(size_t)m * N + n] = (_Float16)v;
            }
        }
    }
}

// ---- windowed attention: scores, softmax, weighted V sum -----------------
// One 256-thread block per (b,s).  Out-of-range neighbors carry score 0 and
// V=0 (the reference's zero-padding), and ARE included in the softmax.
__global__ __launch_bounds__(256) void atten_kernel(
    const float* __restrict__ Q, const float* __restrict__ Km,
    const float* __restrict__ V,
    float* __restrict__ aw, _Float16* __restrict__ awh)
{
    __shared__ float red[256];
    __shared__ float probs[WN];

    const int row = blockIdx.x;         // 0..NROWS-1
    const int b   = row >> 11;          // / SEQ
    const int s   = row & (SEQ - 1);
    const int tid = threadIdx.x;

    float part[WN];
#pragma unroll
    for (int w = 0; w < WN; ++w) part[w] = 0.0f;

    for (int d = tid; d < HID_DIM; d += 256) {
        float qv = Q[(size_t)row * HID_DIM + d];
#pragma unroll
        for (int w = 0; w < WN; ++w) {
            int sn = s + ASZ - w;
            if (sn >= 0 && sn < SEQ)
                part[w] += qv * Km[(size_t)(b * SEQ + sn) * HID_DIM + d];
        }
    }

    const float rscale = 0.0441941738241592f; // 1/sqrt(512)
    for (int w = 0; w < WN; ++w) {
        red[tid] = part[w];
        __syncthreads();
        for (int off = 128; off > 0; off >>= 1) {
            if (tid < off) red[tid] += red[tid + off];
            __syncthreads();
        }
        if (tid == 0) probs[w] = red[0] * rscale;
        __syncthreads();
    }

    if (tid == 0) {
        float mx = probs[0];
        for (int w = 1; w < WN; ++w) mx = fmaxf(mx, probs[w]);
        float sum = 0.0f;
        for (int w = 0; w < WN; ++w) { probs[w] = expf(probs[w] - mx); sum += probs[w]; }
        float inv = 1.0f / sum;
        for (int w = 0; w < WN; ++w) probs[w] *= inv;
    }
    __syncthreads();

    float p[WN];
#pragma unroll
    for (int w = 0; w < WN; ++w) p[w] = probs[w];

    for (int d = tid; d < HID_DIM; d += 256) {
        float acc = 0.0f;
#pragma unroll
        for (int w = 0; w < WN; ++w) {
            int sn = s + ASZ - w;
            if (sn >= 0 && sn < SEQ)
                acc += p[w] * V[(size_t)(b * SEQ + sn) * HID_DIM + d];
        }
        aw[(size_t)row * HID_DIM + d]  = acc;
        awh[(size_t)row * HID_DIM + d] = (_Float16)acc;
    }
}

// ---- final tiny FC: out[row,o] = bfc[o] + sum_d g[row,d]*Wfc[d,o] --------
__global__ void final_fc_kernel(const _Float16* __restrict__ G,
                                const float* __restrict__ Wfc,
                                const float* __restrict__ bfc,
                                float* __restrict__ out)
{
    int idx = blockIdx.x * blockDim.x + threadIdx.x;  // NROWS*2
    if (idx >= NROWS * 2) return;
    int row = idx >> 1;
    int o   = idx & 1;
    float acc = bfc[o];
    const _Float16* g = G + (size_t)row * HID_DIM;
    for (int d = 0; d < HID_DIM; ++d) acc += (float)g[d] * Wfc[d * 2 + o];
    out[idx] = acc;
}

// ---------------------------------------------------------------------------
extern "C" void kernel_launch(void* const* d_in, const int* in_sizes, int n_in,
                              void* d_out, int out_size, void* d_ws, size_t ws_size,
                              hipStream_t stream)
{
    (void)in_sizes; (void)n_in; (void)out_size; (void)ws_size;

    const float* x   = (const float*)d_in[0];
    const float* W1  = (const float*)d_in[1];
    const float* b1  = (const float*)d_in[2];
    const float* Wq  = (const float*)d_in[3];
    const float* Wk  = (const float*)d_in[4];
    const float* Wv  = (const float*)d_in[5];
    const float* W2  = (const float*)d_in[6];
    const float* b2  = (const float*)d_in[7];
    const float* Wfc = (const float*)d_in[8];
    const float* bfc = (const float*)d_in[9];

    float* out = (float*)d_out;                  // [16384, 2]
    float* aw  = (float*)d_out + (size_t)NROWS * 2;  // [16384, 512]

    char* ws = (char*)d_ws;
    size_t off = 0;
    auto alloc = [&](size_t bytes) -> void* {
        void* p = ws + off;
        off += (bytes + 255) & ~(size_t)255;
        return p;
    };
    _Float16* xh  = (_Float16*)alloc((size_t)NROWS * HID_DIM * 2);
    _Float16* hh  = (_Float16*)alloc((size_t)NROWS * HID_DIM * 2);
    _Float16* gh  = (_Float16*)alloc((size_t)NROWS * HID_DIM * 2);
    _Float16* W1T = (_Float16*)alloc((size_t)HID_DIM * HID_DIM * 2);
    _Float16* WqT = (_Float16*)alloc((size_t)HID_DIM * HID_DIM * 2);
    _Float16* WkT = (_Float16*)alloc((size_t)HID_DIM * HID_DIM * 2);
    _Float16* WvT = (_Float16*)alloc((size_t)HID_DIM * HID_DIM * 2);
    _Float16* W2T = (_Float16*)alloc((size_t)HID_DIM * HID_DIM * 2);
    float* Qb = (float*)alloc((size_t)NROWS * HID_DIM * 4);
    float* Kb = (float*)alloc((size_t)NROWS * HID_DIM * 4);
    float* Vb = (float*)alloc((size_t)NROWS * HID_DIM * 4);
    _Float16* awh = xh;  // xh is dead after the first GEMM -> reuse

    // 1) convert activations
    int nx = NROWS * HID_DIM;
    cvt_f16_kernel<<<(nx + 255) / 256, 256, 0, stream>>>(x, xh, nx);

    // 2) transpose + convert weights
    int nw = HID_DIM * HID_DIM;
    int twb = (nw + 255) / 256;
    transpose_f16_kernel<<<twb, 256, 0, stream>>>(W1, W1T, HID_DIM, HID_DIM);
    transpose_f16_kernel<<<twb, 256, 0, stream>>>(Wq, WqT, HID_DIM, HID_DIM);
    transpose_f16_kernel<<<twb, 256, 0, stream>>>(Wk, WkT, HID_DIM, HID_DIM);
    transpose_f16_kernel<<<twb, 256, 0, stream>>>(Wv, WvT, HID_DIM, HID_DIM);
    transpose_f16_kernel<<<twb, 256, 0, stream>>>(W2, W2T, HID_DIM, HID_DIM);

    dim3 ggrid(HID_DIM / 32, NROWS / 64);

    // 3) h = relu(x @ W1 + b1)  -> f16
    gemm_wmma_kernel<<<ggrid, 32, 0, stream>>>(xh, W1T, b1, nullptr, hh, 1);

    // 4) Q, K, V = h @ {Wq, Wk, Wv} -> f32
    gemm_wmma_kernel<<<ggrid, 32, 0, stream>>>(hh, WqT, nullptr, Qb, nullptr, 0);
    gemm_wmma_kernel<<<ggrid, 32, 0, stream>>>(hh, WkT, nullptr, Kb, nullptr, 0);
    gemm_wmma_kernel<<<ggrid, 32, 0, stream>>>(hh, WvT, nullptr, Vb, nullptr, 0);

    // 5) windowed attention -> aw (output #2) + f16 copy
    atten_kernel<<<NROWS, 256, 0, stream>>>(Qb, Kb, Vb, aw, awh);

    // 6) g = relu(aw @ W2 + b2) -> f16
    gemm_wmma_kernel<<<ggrid, 32, 0, stream>>>(awh, W2T, b2, nullptr, gh, 1);

    // 7) out = g @ Wfc + bfc (output #1)
    final_fc_kernel<<<(NROWS * 2 + 255) / 256, 256, 0, stream>>>(gh, Wfc, bfc, out);
}
